// Autoformer_76476187673109
// MI455X (gfx1250) — matrix operand order repaired and checked by reading, not compiled, not run on hardware
//
#include <hip/hip_runtime.h>
#include <math.h>

// ---------------------------------------------------------------------------
// Autoformer forward for gfx1250 (MI455X). wave32. WMMA bf16 GEMMs.
//
// Input flattening assumption (setup_inputs dict order, nested dicts in
// insertion order, lists in order, lin = {w,b}):
//   0 src(B,L,128) 1 seasonal_init(B,L,128) 2 trend_init(B,L,32)
//   enc0 @3 : qw qb kw kb vw vb ow ob fc1w fc1b fc2w fc2b ln_g ln_b   (14)
//   enc1 @17: same                                                    (14)
//   dec0 @31: self qw..ob (8) cross qw..ob (8) fc1w fc1b fc2w fc2b
//             ln_g ln_b conv_w(32,128,3)                              (23)
//   54 seasonal_proj.w(32,128)  55 seasonal_proj.b(32)  56 res_conv_w(32,32,3)
// ---------------------------------------------------------------------------

#define BB 4
#define LL 4096
#define CE 128          // EMBED
#define TOPK 8          // int(log(4096)) = 8
#define FFT_LOG 12
#define MM (BB * LL)    // 16384 rows

typedef __attribute__((ext_vector_type(16))) __bf16 v16bf;
typedef __attribute__((ext_vector_type(8)))  float  v8f;

// ============================ WMMA GEMM ====================================
// Y[M,N] = X[M,K] @ W[N,K]^T + bias[N]; act=1 -> exact GELU.
// One wave computes a 16 x (NSUB*16) strip: the A fragment is loaded once per
// K-step and reused by NSUB WMMAs. NSUB is a compile-time constant so there
// is NO control flow around the WMMAs (EXEC stays all-ones, no saveexec, no
// accumulator register shuffling) -- accumulators live in fixed VGPRs.
template <int NSUB>
__global__ __launch_bounds__(256) void gemm_wmma(
    const float* __restrict__ X, const float* __restrict__ W,
    const float* __restrict__ bias, float* __restrict__ Y,
    int M, int N, int K, int act)
{
  const int wave = threadIdx.x >> 5;
  const int lane = threadIdx.x & 31;
  const int tilesN4 = N / (16 * NSUB);           // strips per row block
  const int tilesTot = (M >> 4) * tilesN4;
  const int tile = blockIdx.x * 8 + wave;
  if (tile >= tilesTot) return;                  // wave-uniform
  const int tm = tile / tilesN4, tn4 = tile - tm * tilesN4;
  const int half = lane >> 4;
  const int l15  = lane & 15;
  const float* xrow = X + (size_t)(tm * 16 + l15) * K;
  // B[k][n] = W[n][k]
  const float* wrow[NSUB];
  #pragma unroll
  for (int s = 0; s < NSUB; ++s)
    wrow[s] = W + (size_t)(tn4 * (16 * NSUB) + s * 16 + l15) * K;

  v8f acc[NSUB];
  #pragma unroll
  for (int s = 0; s < NSUB; ++s) acc[s] = (v8f){};

  for (int kk = 0; kk < K; kk += 32) {
    v16bf a;
    const int ka = kk + half * 16;     // A: lanes0-15 K0-15, lanes16-31 K16-31
    #pragma unroll
    for (int j = 0; j < 8; ++j) {
      a[j]     = (__bf16)xrow[ka + j];
      a[j + 8] = (__bf16)xrow[ka + 8 + j];
    }
    __builtin_prefetch(xrow + ka + 32, 0, 1);    // next K-step (global_prefetch_b8)
    const int kb = kk + half * 16;     // B: lanes0-15 K0-15, lanes16-31 K16-31
    #pragma unroll
    for (int s = 0; s < NSUB; ++s) {
      v16bf b;
      #pragma unroll
      for (int e = 0; e < 16; ++e) b[e] = (__bf16)wrow[s][kb + e];
      acc[s] = __builtin_amdgcn_wmma_f32_16x16x32_bf16(false, a, false, b,
                                                       (short)0, acc[s], false, false);
    }
  }
  #pragma unroll
  for (int s = 0; s < NSUB; ++s) {
    const int ncol = tn4 * (16 * NSUB) + s * 16 + l15;
    const float bv = bias[ncol];
    #pragma unroll
    for (int r = 0; r < 8; ++r) {
      const int mo = tm * 16 + half * 8 + r;     // C/D layout
      float v = acc[s][r] + bv;
      if (act) v = 0.5f * v * (1.0f + erff(v * 0.70710678118654752f));
      Y[(size_t)mo * N + ncol] = v;
    }
  }
}

// ============================ FFT autocorrelation ==========================
__device__ __forceinline__ void fft_core(float* re, float* im, int tid, float sgn)
{
  #pragma unroll 1
  for (int s = 1; s <= FFT_LOG; ++s) {
    const int half = 1 << (s - 1);
    const float w0 = sgn * 6.283185307179586f / (float)(1 << s);
    for (int j = tid; j < (1 << (FFT_LOG - 1)); j += 256) {
      const int grp = j >> (s - 1);
      const int pos = j & (half - 1);
      const int i0 = (grp << s) + pos;
      const int i1 = i0 + half;
      float sn, cs;
      __sincosf(w0 * (float)pos, &sn, &cs);
      const float xr = re[i1], xi = im[i1];
      const float tr = cs * xr - sn * xi;
      const float ti = cs * xi + sn * xr;
      const float ur = re[i0], ui = im[i0];
      re[i0] = ur + tr; im[i0] = ui + ti;
      re[i1] = ur - tr; im[i1] = ui - ti;
    }
    __syncthreads();
  }
}

// corr[row, tau] = irfft( rfft(q_row) * conj(rfft(k_row)) ), row = b*128 + h*16 + d
__global__ __launch_bounds__(256) void corr_kernel(
    const float* __restrict__ Q, const float* __restrict__ Km,
    float* __restrict__ corr)
{
  __shared__ float re[1 << FFT_LOG];
  __shared__ float im[1 << FFT_LOG];
  const int row = blockIdx.x;
  const int b  = row >> 7;
  const int hd = row & 127;
  const float* qp = Q  + ((size_t)b * LL) * CE + hd;
  const float* kp = Km + ((size_t)b * LL) * CE + hd;
  const int tid = threadIdx.x;
  const int N = 1 << FFT_LOG;
  // bit-reversed load, z = q + i*k (one FFT for both real signals)
  for (int t = tid; t < N; t += 256) {
    const int r = (int)(__brev((unsigned)t) >> (32 - FFT_LOG));
    re[r] = qp[(size_t)t * CE];
    im[r] = kp[(size_t)t * CE];
  }
  __syncthreads();
  fft_core(re, im, tid, -1.0f);                  // forward
  // spectral unpack + P = Fq * conj(Fk); pairs (f, N-f) owned by one thread
  for (int f = tid; f <= N / 2; f += 256) {
    if (f == 0) {
      re[0] = re[0] * im[0];                     // Fq0*Fk0, both real
      im[0] = 0.0f;
    } else {
      const int g = N - f;                       // f==N/2 -> g==f (self-pair)
      const float ar = re[f], ai = im[f];
      const float br = re[g], bi = -im[g];       // conj(Z[N-f])
      const float fqr = 0.5f * (ar + br), fqi = 0.5f * (ai + bi);
      const float dr = ar - br, di = ai - bi;    // Fk = -0.5i*(a-b)
      const float fkr = 0.5f * di, fki = -0.5f * dr;
      const float pr = fqr * fkr + fqi * fki;    // Fq * conj(Fk)
      const float pi = fqi * fkr - fqr * fki;
      re[f] = pr; im[f] = pi;
      re[g] = pr; im[g] = -pi;                   // conj symmetry
    }
  }
  __syncthreads();
  // bit-reverse permute for the inverse pass
  for (int t = tid; t < N; t += 256) {
    const int r = (int)(__brev((unsigned)t) >> (32 - FFT_LOG));
    if (r > t) {
      float x = re[t]; re[t] = re[r]; re[r] = x;
      float y = im[t]; im[t] = im[r]; im[r] = y;
    }
  }
  __syncthreads();
  fft_core(re, im, tid, 1.0f);                   // inverse (unscaled)
  const float invN = 1.0f / (float)N;
  float* cp = corr + (size_t)row * LL;
  for (int t = tid; t < N; t += 256) cp[t] = re[t] * invN;
}

// mean over (h,d) -> mean_value[b, tau]
__global__ void meancorr_kernel(const float* __restrict__ corr, float* __restrict__ mv)
{
  int i = blockIdx.x * blockDim.x + threadIdx.x;
  if (i >= BB * LL) return;
  int b = i / LL, t = i - b * LL;
  const float* base = corr + (size_t)b * CE * LL + t;
  float s = 0.f;
  for (int hd = 0; hd < CE; ++hd) s += base[(size_t)hd * LL];
  mv[i] = s * (1.0f / CE);
}

__global__ void gmean_kernel(const float* __restrict__ mv, float* __restrict__ gm)
{
  int t = blockIdx.x * blockDim.x + threadIdx.x;
  if (t >= LL) return;
  float s = 0.f;
  for (int b = 0; b < BB; ++b) s += mv[(size_t)b * LL + t];
  gm[t] = s * (1.0f / BB);
}

// single block: top-8 lags of gm, then per-batch softmax of mv at those lags
__global__ __launch_bounds__(256) void topk_softmax_kernel(
    const float* __restrict__ gm, const float* __restrict__ mv,
    int* __restrict__ idx_out, float* __restrict__ w_out)
{
  __shared__ float ls[LL];
  __shared__ float rv[256];
  __shared__ int   ri[256];
  __shared__ int   sidx[TOPK];
  const int tid = threadIdx.x;
  for (int t = tid; t < LL; t += 256) ls[t] = gm[t];
  __syncthreads();
  for (int it = 0; it < TOPK; ++it) {
    float best = -3.4e38f; int bi = 0;
    for (int t = tid; t < LL; t += 256) {
      float v = ls[t];
      if (v > best) { best = v; bi = t; }
    }
    rv[tid] = best; ri[tid] = bi;
    __syncthreads();
    for (int off = 128; off > 0; off >>= 1) {
      if (tid < off && rv[tid + off] > rv[tid]) { rv[tid] = rv[tid + off]; ri[tid] = ri[tid + off]; }
      __syncthreads();
    }
    if (tid == 0) { sidx[it] = ri[0]; ls[ri[0]] = -3.4e38f; }
    __syncthreads();
  }
  if (tid < BB) {
    float vals[TOPK]; float m = -3.4e38f;
    for (int i = 0; i < TOPK; ++i) { vals[i] = mv[(size_t)tid * LL + sidx[i]]; m = fmaxf(m, vals[i]); }
    float s = 0.f;
    for (int i = 0; i < TOPK; ++i) { vals[i] = __expf(vals[i] - m); s += vals[i]; }
    const float inv = 1.0f / s;
    for (int i = 0; i < TOPK; ++i) w_out[tid * TOPK + i] = vals[i] * inv;
  }
  if (tid < TOPK) idx_out[tid] = sidx[tid];
}

// agg[b,t,c] = sum_i w[b,i] * V[b, (t+idx[i]) % L, c]
__global__ void agg_kernel(const float* __restrict__ V, const int* __restrict__ idx,
                           const float* __restrict__ w, float* __restrict__ out, int total)
{
  int i = blockIdx.x * blockDim.x + threadIdx.x;
  if (i >= total) return;
  const int c = i & (CE - 1);
  const int t = (i >> 7) & (LL - 1);
  const int b = i >> 19;                 // / (LL*CE)
  const float* vb = V + ((size_t)b << 19);
  float s = 0.f;
  #pragma unroll
  for (int q = 0; q < TOPK; ++q) {
    const int sh = (t + idx[q]) & (LL - 1);
    s += w[b * TOPK + q] * vb[((size_t)sh << 7) + c];
  }
  out[i] = s;
}

// ============================ series_decomp (C=128) ========================
// seasonal = x - movavg25(x, edge-replicate); mode 1: trend=tr, 2: trend+=tr
__global__ void decomp_kernel(const float* __restrict__ x, float* __restrict__ seas,
                              float* __restrict__ trend, int total, int mode)
{
  int i = blockIdx.x * blockDim.x + threadIdx.x;
  if (i >= total) return;
  const int c = i & (CE - 1);
  const int t = (i >> 7) & (LL - 1);
  const int b = i >> 19;
  const float* xb = x + ((size_t)b << 19) + c;
  float sum = 0.f;
  #pragma unroll 1
  for (int j = -12; j <= 12; ++j) {
    int tt = t + j;
    tt = tt < 0 ? 0 : (tt > LL - 1 ? LL - 1 : tt);
    sum += xb[(size_t)tt << 7];
  }
  const float tr = sum * (1.0f / 25.0f);
  seas[i] = x[i] - tr;
  if (mode == 1)      trend[i]  = tr;
  else if (mode == 2) trend[i] += tr;
}

// ============================ layer_norm_special ===========================
__global__ __launch_bounds__(128) void ln_row_kernel(
    const float* __restrict__ x, const float* __restrict__ g,
    const float* __restrict__ bta, float* __restrict__ xh)
{
  __shared__ float s[CE];
  const int row = blockIdx.x;
  const int c = threadIdx.x;
  const float v = x[(size_t)row * CE + c];
  s[c] = v; __syncthreads();
  for (int off = 64; off > 0; off >>= 1) { if (c < off) s[c] += s[c + off]; __syncthreads(); }
  const float mu = s[0] * (1.0f / CE);
  __syncthreads();
  const float d = v - mu;
  s[c] = d * d; __syncthreads();
  for (int off = 64; off > 0; off >>= 1) { if (c < off) s[c] += s[c + off]; __syncthreads(); }
  const float var = s[0] * (1.0f / CE);
  xh[(size_t)row * CE + c] = d * rsqrtf(var + 1e-5f) * g[c] + bta[c];
}

__global__ __launch_bounds__(256) void colmean_kernel(
    const float* __restrict__ xh, float* __restrict__ cm)
{
  __shared__ float s[256];
  const int col = blockIdx.x;          // b*CE + c
  const int b = col >> 7, c = col & 127;
  const float* base = xh + ((size_t)b * LL) * CE + c;
  float sum = 0.f;
  for (int t = threadIdx.x; t < LL; t += 256) sum += base[(size_t)t * CE];
  s[threadIdx.x] = sum; __syncthreads();
  for (int off = 128; off > 0; off >>= 1) {
    if (threadIdx.x < off) s[threadIdx.x] += s[threadIdx.x + off];
    __syncthreads();
  }
  if (threadIdx.x == 0) cm[col] = s[0] * (1.0f / LL);
}

__global__ void sub_colmean_kernel(const float* __restrict__ xh, const float* __restrict__ cm,
                                   float* __restrict__ y, int total)
{
  int i = blockIdx.x * blockDim.x + threadIdx.x;
  if (i >= total) return;
  const int c = i & 127;
  const int b = i >> 19;
  y[i] = xh[i] - cm[b * CE + c];
}

// ============================ circular conv (k=3) ==========================
__global__ void circconv_kernel(const float* __restrict__ x, const float* __restrict__ w,
                                float* __restrict__ y, int Cin, int Cout, int total, int accum)
{
  int i = blockIdx.x * blockDim.x + threadIdx.x;
  if (i >= total) return;
  const int co = i % Cout;
  const int t  = (i / Cout) % LL;
  const int b  = i / (Cout * LL);
  const float* xb = x + (size_t)b * LL * Cin;
  float s = 0.f;
  for (int j = 0; j < 3; ++j) {
    int tt = t - 1 + j;
    tt = tt < 0 ? tt + LL : (tt >= LL ? tt - LL : tt);
    const float* xr = xb + (size_t)tt * Cin;
    const float* wr = w + ((size_t)co * Cin) * 3 + j;   // w[co][ci][j]
    for (int ci = 0; ci < Cin; ++ci) s += xr[ci] * wr[ci * 3];
  }
  if (accum) y[i] += s; else y[i] = s;
}

__global__ void add_kernel(const float* __restrict__ a, const float* __restrict__ b,
                           float* __restrict__ y, int n)
{
  int i = blockIdx.x * blockDim.x + threadIdx.x;
  if (i < n) y[i] = a[i] + b[i];
}

// ============================ Orchestration ================================
extern "C" void kernel_launch(void* const* d_in, const int* in_sizes, int n_in,
                              void* d_out, int out_size, void* d_ws, size_t ws_size,
                              hipStream_t stream)
{
  (void)in_sizes; (void)n_in; (void)out_size; (void)ws_size;
  const size_t NXL = (size_t)MM * CE;            // 2,097,152 floats (8 MB)
  float* ws = (float*)d_ws;
  size_t off = 0;
  float* X    = ws + off; off += NXL;            // current state
  float* Qb   = ws + off; off += NXL;            // q proj / residual sum
  float* TMP  = ws + off; off += NXL;            // attn/ffn output
  float* Kb   = ws + off; off += NXL;            // k proj  (FFN aliases Kb..+4NXL)
  float* Vb   = ws + off; off += NXL;            // v proj
  float* CORR = ws + off; off += NXL;            // corr, reused as agg
  float* TMP2 = ws + off; off += NXL;            // LN xh (part of FFN alias)
  float* TSUM = ws + off; off += NXL;            // decoder trend accumulator
  float* ENC  = ws + off; off += NXL;            // encoder output
  float* FFN  = Kb;                              // (B,L,512) aliases Kb..TMP2
  float* TREND = ws + off; off += (size_t)MM * 32;
  float* SEAS  = ws + off; off += (size_t)MM * 32;
  float* MV    = ws + off; off += (size_t)BB * LL;
  float* GM    = ws + off; off += LL;
  float* WTS   = ws + off; off += 64;
  int*   IDX   = (int*)(ws + off); off += 64;
  float* CM    = ws + off; off += (size_t)BB * CE;

  auto F = [&](int i) { return (const float*)d_in[i]; };
  const int nTot = (int)NXL;
  const int EB = 256;

  auto gemm = [&](const float* Xp, int wi, int bi, float* Yp, int N, int K, int act) {
    if (N % 64 == 0) {
      const int tiles = (MM / 16) * (N / 64);
      gemm_wmma<4><<<dim3((tiles + 7) / 8), dim3(256), 0, stream>>>(Xp, F(wi), F(bi), Yp, MM, N, K, act);
    } else {           // N == 32
      const int tiles = (MM / 16) * (N / 32);
      gemm_wmma<2><<<dim3((tiles + 7) / 8), dim3(256), 0, stream>>>(Xp, F(wi), F(bi), Yp, MM, N, K, act);
    }
  };
  auto eadd = [&](const float* a, const float* b, float* y, int n) {
    add_kernel<<<dim3((n + EB - 1) / EB), dim3(EB), 0, stream>>>(a, b, y, n);
  };
  auto decomp = [&](const float* x, float* seas, float* tr, int mode) {
    decomp_kernel<<<dim3((nTot + EB - 1) / EB), dim3(EB), 0, stream>>>(x, seas, tr, nTot, mode);
  };
  auto attention = [&](const float* qin, const float* kvin, int base, float* outp) {
    gemm(qin,  base + 0, base + 1, Qb, CE, CE, 0);
    gemm(kvin, base + 2, base + 3, Kb, CE, CE, 0);
    gemm(kvin, base + 4, base + 5, Vb, CE, CE, 0);
    corr_kernel<<<dim3(BB * CE), dim3(256), 0, stream>>>(Qb, Kb, CORR);
    meancorr_kernel<<<dim3((BB * LL + EB - 1) / EB), dim3(EB), 0, stream>>>(CORR, MV);
    gmean_kernel<<<dim3((LL + EB - 1) / EB), dim3(EB), 0, stream>>>(MV, GM);
    topk_softmax_kernel<<<dim3(1), dim3(256), 0, stream>>>(GM, MV, IDX, WTS);
    agg_kernel<<<dim3((nTot + EB - 1) / EB), dim3(EB), 0, stream>>>(Vb, IDX, WTS, CORR, nTot);
    gemm(CORR, base + 6, base + 7, outp, CE, CE, 0);
  };
  auto lnorm = [&](int gi, int bi) {
    ln_row_kernel<<<dim3(MM), dim3(128), 0, stream>>>(X, F(gi), F(bi), TMP2);
    colmean_kernel<<<dim3(BB * CE), dim3(256), 0, stream>>>(TMP2, CM);
    sub_colmean_kernel<<<dim3((nTot + EB - 1) / EB), dim3(EB), 0, stream>>>(TMP2, CM, X, nTot);
  };

  // ---------------- Encoder ----------------
  hipMemcpyAsync(X, d_in[0], NXL * sizeof(float), hipMemcpyDeviceToDevice, stream);
  for (int e = 0; e < 2; ++e) {
    const int eb = 3 + e * 14;
    attention(X, X, eb, TMP);
    eadd(X, TMP, Qb, nTot);
    decomp(Qb, X, nullptr, 0);
    gemm(X, eb + 8, eb + 9, FFN, 512, CE, 1);    // GELU
    gemm(FFN, eb + 10, eb + 11, TMP, CE, 512, 0);
    eadd(X, TMP, Qb, nTot);
    decomp(Qb, X, nullptr, 0);
    lnorm(eb + 12, eb + 13);
  }
  hipMemcpyAsync(ENC, X, NXL * sizeof(float), hipMemcpyDeviceToDevice, stream);

  // ---------------- Decoder ----------------
  hipMemcpyAsync(X, d_in[1], NXL * sizeof(float), hipMemcpyDeviceToDevice, stream);
  attention(X, X, 31, TMP);                      // self
  eadd(X, TMP, Qb, nTot);
  decomp(Qb, X, TSUM, 1);                        // t1 (overwrite)
  attention(X, ENC, 39, TMP);                    // cross
  eadd(X, TMP, Qb, nTot);
  decomp(Qb, X, TSUM, 2);                        // + t2
  gemm(X, 47, 48, FFN, 512, CE, 1);
  gemm(FFN, 49, 50, TMP, CE, 512, 0);
  eadd(X, TMP, Qb, nTot);
  decomp(Qb, X, TSUM, 2);                        // + t3
  lnorm(51, 52);

  // trend path: res_conv(trend_init) + conv(t1+t2+t3)
  const int n32 = MM * 32;
  circconv_kernel<<<dim3((n32 + EB - 1) / EB), dim3(EB), 0, stream>>>(F(2), F(56), TREND, 32, 32, n32, 0);
  circconv_kernel<<<dim3((n32 + EB - 1) / EB), dim3(EB), 0, stream>>>(TSUM, F(53), TREND, CE, 32, n32, 1);

  // seasonal projection + final sum
  gemm(X, 54, 55, SEAS, 32, CE, 0);
  eadd(SEAS, TREND, (float*)d_out, n32);
}
